// APPNP_4148938408472
// MI455X (gfx1250) — matrix operand order repaired and checked by reading, not compiled, not run on hardware
//
#include <hip/hip_runtime.h>
#include <math.h>

#define N_NODES 100000
#define E_EDGES 3200000
#define NFEAT   512
#define HID     64
#define NCLASS  64
#define ALPHA   0.1f
#define NITER   10

typedef __bf16 bf16;
typedef __attribute__((ext_vector_type(16))) __bf16       v16bf;
typedef __attribute__((ext_vector_type(8)))  float        v8f;
typedef __attribute__((ext_vector_type(4)))  float        f32x4;
typedef __attribute__((ext_vector_type(4)))  unsigned int u32x4;

union BFrag {
  u32x4 q[2];
  v16bf v;
};

// ---------------------------------------------------------------------------
// GEMM1: h[N,64] = relu(x[N,512] @ W1[512,64]) ; bf16 WMMA, f32 accumulate.
// Block = 128 threads (4 waves), 64 rows per block, W1^T staged in LDS (bf16,
// 16-halfword-block XOR swizzle to spread LDS banks on B-fragment loads).
// x is streamed once (205 MB): non-temporal loads + prefetch keep it from
// rinsing L2, which the propagation phase wants for z/edges.
// ---------------------------------------------------------------------------
__launch_bounds__(128)
__global__ void appnp_gemm1_relu(const float* __restrict__ x,
                                 const float* __restrict__ W1,
                                 bf16* __restrict__ h) {
  __shared__ bf16 sW[HID * NFEAT];  // 64*512*2 = 64 KB, layout [n][k-swizzled]

  const int tid = threadIdx.x;
  // Cooperative transposed fill: W1[k][n] -> sW[n][swz(k)]
  for (int i = tid; i < NFEAT * HID; i += 128) {
    int k = i >> 6;          // 0..511
    int n = i & (HID - 1);   // 0..63
    int kswz = ((((k >> 4) ^ (n & 7)) << 4) | (k & 15));
    sW[n * NFEAT + kswz] = (bf16)W1[i];
  }
  __syncthreads();

  const int wave = tid >> 5;
  const int lane = tid & 31;
  const int m    = lane & 15;
  const int hi   = lane >> 4;   // 0 or 1
  const int kbl  = hi << 4;     // A-fragment K sub-base: 0 or 16

  const int rowA = blockIdx.x * 64 + wave * 16 + m;
  const int rowc = rowA < N_NODES ? rowA : (N_NODES - 1);  // clamp (EXEC all-1 for WMMA)
  const float* __restrict__ xrow = x + (size_t)rowc * NFEAT;

  v8f c0 = {}, c1 = {}, c2 = {}, c3 = {};

  for (int k0 = 0; k0 < NFEAT; k0 += 32) {
    // Prefetch the x stream 4 k-steps (512 B/row) ahead (speculative; OOB
    // prefetches are dropped by hardware).
    __builtin_prefetch(xrow + k0 + 128, 0, 0);
    // ---- A fragment (16x32 bf16): lane holds K = k0+kbl+[0..15]
    const f32x4* pa = (const f32x4*)(xrow + k0 + kbl);
    f32x4 a0 = __builtin_nontemporal_load(pa + 0);   // K    .. K+3
    f32x4 a1 = __builtin_nontemporal_load(pa + 1);   // K+4  .. K+7
    f32x4 a2 = __builtin_nontemporal_load(pa + 2);   // K+8  .. K+11
    f32x4 a3 = __builtin_nontemporal_load(pa + 3);   // K+12 .. K+15
    v16bf a;
#pragma unroll
    for (int i = 0; i < 4; ++i) {
      a[i]      = (bf16)a0[i];
      a[4 + i]  = (bf16)a1[i];
      a[8 + i]  = (bf16)a2[i];
      a[12 + i] = (bf16)a3[i];
    }
    // ---- B fragments (32x16 bf16): lane = column, K = k0 + hi*16 + [0..15]
    const int koffB = k0 + (hi << 4);
    const int swz   = (((koffB >> 4) ^ (m & 7)) << 4);
    BFrag b0, b1, b2, b3;
    {
      const u32x4* pb = (const u32x4*)&sW[(0 * 16 + m) * NFEAT + swz];
      b0.q[0] = pb[0]; b0.q[1] = pb[1];
    }
    {
      const u32x4* pb = (const u32x4*)&sW[(1 * 16 + m) * NFEAT + swz];
      b1.q[0] = pb[0]; b1.q[1] = pb[1];
    }
    {
      const u32x4* pb = (const u32x4*)&sW[(2 * 16 + m) * NFEAT + swz];
      b2.q[0] = pb[0]; b2.q[1] = pb[1];
    }
    {
      const u32x4* pb = (const u32x4*)&sW[(3 * 16 + m) * NFEAT + swz];
      b3.q[0] = pb[0]; b3.q[1] = pb[1];
    }
    c0 = __builtin_amdgcn_wmma_f32_16x16x32_bf16(false, a, false, b0.v, (short)0, c0, false, false);
    c1 = __builtin_amdgcn_wmma_f32_16x16x32_bf16(false, a, false, b1.v, (short)0, c1, false, false);
    c2 = __builtin_amdgcn_wmma_f32_16x16x32_bf16(false, a, false, b2.v, (short)0, c2, false, false);
    c3 = __builtin_amdgcn_wmma_f32_16x16x32_bf16(false, a, false, b3.v, (short)0, c3, false, false);
  }

  // C/D layout: VGPR r -> M = r (lanes 0-15) / 8+r (lanes 16-31); N = lane&15
  const int orow_base = blockIdx.x * 64 + wave * 16 + (hi << 3);
#pragma unroll
  for (int r = 0; r < 8; ++r) {
    const int orow = orow_base + r;
    if (orow < N_NODES) {
      bf16* hrow = h + (size_t)orow * HID;
      hrow[0 * 16 + m] = (bf16)fmaxf(c0[r], 0.0f);
      hrow[1 * 16 + m] = (bf16)fmaxf(c1[r], 0.0f);
      hrow[2 * 16 + m] = (bf16)fmaxf(c2[r], 0.0f);
      hrow[3 * 16 + m] = (bf16)fmaxf(c3[r], 0.0f);
    }
  }
}

// ---------------------------------------------------------------------------
// GEMM2: z0[N,64] = h[N,64] @ W2[64,64]; also z_cur = z0 (APPNP init).
// ---------------------------------------------------------------------------
__launch_bounds__(128)
__global__ void appnp_gemm2(const bf16* __restrict__ h,
                            const float* __restrict__ W2,
                            float* __restrict__ z0,
                            float* __restrict__ zcur) {
  __shared__ bf16 sW[HID * HID];  // 8 KB, [n][k-swizzled]

  const int tid = threadIdx.x;
  for (int i = tid; i < HID * HID; i += 128) {
    int k = i >> 6;
    int n = i & (HID - 1);
    int kswz = ((((k >> 4) ^ (n & 3)) << 4) | (k & 15));
    sW[n * HID + kswz] = (bf16)W2[i];
  }
  __syncthreads();

  const int wave = tid >> 5;
  const int lane = tid & 31;
  const int m    = lane & 15;
  const int hi   = lane >> 4;
  const int kbl  = hi << 4;

  const int rowA = blockIdx.x * 64 + wave * 16 + m;
  const int rowc = rowA < N_NODES ? rowA : (N_NODES - 1);
  const bf16* __restrict__ hrow = h + (size_t)rowc * HID;

  v8f c0 = {}, c1 = {}, c2 = {}, c3 = {};

#pragma unroll
  for (int k0 = 0; k0 < HID; k0 += 32) {
    // A fragment straight from bf16 h: K = k0+kbl+[0..15]
    BFrag a;
    const u32x4* pa = (const u32x4*)(hrow + k0 + kbl);
    a.q[0] = pa[0];   // K .. K+7
    a.q[1] = pa[1];   // K+8 .. K+15
    const int koffB = k0 + (hi << 4);
    const int swz   = (((koffB >> 4) ^ (m & 3)) << 4);
    BFrag b0, b1, b2, b3;
    {
      const u32x4* pb = (const u32x4*)&sW[(0 * 16 + m) * HID + swz];
      b0.q[0] = pb[0]; b0.q[1] = pb[1];
    }
    {
      const u32x4* pb = (const u32x4*)&sW[(1 * 16 + m) * HID + swz];
      b1.q[0] = pb[0]; b1.q[1] = pb[1];
    }
    {
      const u32x4* pb = (const u32x4*)&sW[(2 * 16 + m) * HID + swz];
      b2.q[0] = pb[0]; b2.q[1] = pb[1];
    }
    {
      const u32x4* pb = (const u32x4*)&sW[(3 * 16 + m) * HID + swz];
      b3.q[0] = pb[0]; b3.q[1] = pb[1];
    }
    c0 = __builtin_amdgcn_wmma_f32_16x16x32_bf16(false, a.v, false, b0.v, (short)0, c0, false, false);
    c1 = __builtin_amdgcn_wmma_f32_16x16x32_bf16(false, a.v, false, b1.v, (short)0, c1, false, false);
    c2 = __builtin_amdgcn_wmma_f32_16x16x32_bf16(false, a.v, false, b2.v, (short)0, c2, false, false);
    c3 = __builtin_amdgcn_wmma_f32_16x16x32_bf16(false, a.v, false, b3.v, (short)0, c3, false, false);
  }

  const int orow_base = blockIdx.x * 64 + wave * 16 + (hi << 3);
#pragma unroll
  for (int r = 0; r < 8; ++r) {
    const int orow = orow_base + r;
    if (orow < N_NODES) {
      size_t base = (size_t)orow * NCLASS;
      z0[base + 0 * 16 + m] = c0[r];  zcur[base + 0 * 16 + m] = c0[r];
      z0[base + 1 * 16 + m] = c1[r];  zcur[base + 1 * 16 + m] = c1[r];
      z0[base + 2 * 16 + m] = c2[r];  zcur[base + 2 * 16 + m] = c2[r];
      z0[base + 3 * 16 + m] = c3[r];  zcur[base + 3 * 16 + m] = c3[r];
    }
  }
}

// ---------------------------------------------------------------------------
// znext = ALPHA * z0   (vectorized float4; N*NCLASS/4 = 1.6M exact)
// ---------------------------------------------------------------------------
__launch_bounds__(256)
__global__ void appnp_init_alpha(const float* __restrict__ z0,
                                 float* __restrict__ znext) {
  size_t i = (size_t)blockIdx.x * 256 + threadIdx.x;
  const f32x4* a = (const f32x4*)z0;
  f32x4* b = (f32x4*)znext;
  if (i < (size_t)N_NODES * NCLASS / 4) b[i] = ALPHA * a[i];
}

// ---------------------------------------------------------------------------
// Edge scatter: one wave per edge, lane l owns classes l and l+32.
// znext[dst] += w * z[src]   (f32 global atomics; L2-resident working set)
// ---------------------------------------------------------------------------
__launch_bounds__(256)
__global__ void appnp_scatter(const int* __restrict__ src,
                              const int* __restrict__ dst,
                              const float* __restrict__ w,
                              const float* __restrict__ z,
                              float* __restrict__ znext) {
  const int e = blockIdx.x * 8 + (threadIdx.x >> 5);
  const int lane = threadIdx.x & 31;
  if (e >= E_EDGES) return;
  const int s = src[e];
  const int d = dst[e];
  const float ww = w[e];
  const size_t sb = (size_t)s * NCLASS;
  const size_t db = (size_t)d * NCLASS;
  atomicAdd(&znext[db + lane],      ww * z[sb + lane]);
  atomicAdd(&znext[db + 32 + lane], ww * z[sb + 32 + lane]);
}

// ---------------------------------------------------------------------------
// Row-wise log_softmax: one wave per row, wave32 shuffle reductions.
// Output is written once and never re-read on device -> non-temporal stores.
// ---------------------------------------------------------------------------
__launch_bounds__(256)
__global__ void appnp_log_softmax(const float* __restrict__ z,
                                  float* __restrict__ out) {
  const int row = blockIdx.x * 8 + (threadIdx.x >> 5);
  const int lane = threadIdx.x & 31;
  if (row >= N_NODES) return;
  const size_t b = (size_t)row * NCLASS;
  float v0 = z[b + lane];
  float v1 = z[b + 32 + lane];
  float mx = fmaxf(v0, v1);
#pragma unroll
  for (int off = 16; off > 0; off >>= 1) mx = fmaxf(mx, __shfl_xor(mx, off, 32));
  float s = __expf(v0 - mx) + __expf(v1 - mx);
#pragma unroll
  for (int off = 16; off > 0; off >>= 1) s += __shfl_xor(s, off, 32);
  const float lse = mx + __logf(s);
  __builtin_nontemporal_store(v0 - lse, &out[b + lane]);
  __builtin_nontemporal_store(v1 - lse, &out[b + 32 + lane]);
}

// ---------------------------------------------------------------------------
extern "C" void kernel_launch(void* const* d_in, const int* in_sizes, int n_in,
                              void* d_out, int out_size, void* d_ws, size_t ws_size,
                              hipStream_t stream) {
  (void)in_sizes; (void)n_in; (void)out_size; (void)ws_size;
  const float* x  = (const float*)d_in[0];
  const int*   ei = (const int*)d_in[1];
  const float* ew = (const float*)d_in[2];
  const float* W1 = (const float*)d_in[3];
  const float* W2 = (const float*)d_in[4];
  const int* src = ei;
  const int* dst = ei + E_EDGES;

  // Workspace layout (256B aligned): h(bf16) | z0 | zA | zB
  char* ws = (char*)d_ws;
  size_t off = 0;
  bf16* h = (bf16*)(ws + off);
  off += (size_t)N_NODES * HID * sizeof(bf16);   off = (off + 255) & ~(size_t)255;
  float* z0 = (float*)(ws + off);
  off += (size_t)N_NODES * NCLASS * sizeof(float); off = (off + 255) & ~(size_t)255;
  float* zA = (float*)(ws + off);
  off += (size_t)N_NODES * NCLASS * sizeof(float); off = (off + 255) & ~(size_t)255;
  float* zB = (float*)(ws + off);

  const int row_blocks = (N_NODES + 63) / 64;   // 1563
  appnp_gemm1_relu<<<row_blocks, 128, 0, stream>>>(x, W1, h);
  appnp_gemm2<<<row_blocks, 128, 0, stream>>>(h, W2, z0, zA);

  float* cur = zA;
  float* nxt = zB;
  const int init_blocks = (N_NODES * NCLASS / 4 + 255) / 256;  // 6250
  const int edge_blocks = E_EDGES / 8;                          // 400000
  for (int it = 0; it < NITER; ++it) {
    appnp_init_alpha<<<init_blocks, 256, 0, stream>>>(z0, nxt);
    appnp_scatter<<<edge_blocks, 256, 0, stream>>>(src, dst, ew, cur, nxt);
    float* t = cur; cur = nxt; nxt = t;
  }

  appnp_log_softmax<<<(N_NODES + 7) / 8, 256, 0, stream>>>(cur, (float*)d_out);
}
